// MCBD_63660005261506
// MI455X (gfx1250) — matrix operand order repaired and checked by reading, not compiled, not run on hardware
//
#include <hip/hip_runtime.h>
#include <math.h>

typedef __bf16 bf16_t;
typedef __attribute__((ext_vector_type(16))) __bf16 v16bf;
typedef __attribute__((ext_vector_type(8)))  __bf16 v8bf;
typedef __attribute__((ext_vector_type(4)))  __bf16 v4bf;
typedef __attribute__((ext_vector_type(8)))  float  v8f;

__device__ __forceinline__ float  bf2f(bf16_t v) { return (float)v; }
__device__ __forceinline__ bf16_t f2bf(float v)  { return (bf16_t)v; }

// ---------------------------------------------------------------------------
// 1D conv, kernel=3, pad=1, ReLU.  grid=(Cout,N), block=128 (L)
// ---------------------------------------------------------------------------
__global__ __launch_bounds__(128) void conv1d_k3_relu(
    const float* __restrict__ in, const float* __restrict__ w,
    const float* __restrict__ b, float* __restrict__ out,
    int Cin, int inBatchStride, int Cout)
{
    const int l = threadIdx.x, oc = blockIdx.x, n = blockIdx.y;
    const float* wr = w + (size_t)oc * Cin * 3;
    const float* ib = in + (size_t)n * inBatchStride;
    float acc = b[oc];
    for (int ic = 0; ic < Cin; ++ic) {
        const float* row = ib + ic * 128;
        float x0 = (l > 0)   ? row[l - 1] : 0.f;
        float x1 = row[l];
        float x2 = (l < 127) ? row[l + 1] : 0.f;
        const float* wi = wr + ic * 3;
        acc = fmaf(wi[0], x0, acc);
        acc = fmaf(wi[1], x1, acc);
        acc = fmaf(wi[2], x2, acc);
    }
    out[((size_t)n * Cout + oc) * 128 + l] = fmaxf(acc, 0.f);
}

// ---------------------------------------------------------------------------
// conv3 (1x1, 128->3) + sigmoid(0.01*x) -> tail of d_out.  grid=(3,N), block=128
// ---------------------------------------------------------------------------
__global__ __launch_bounds__(128) void conv3_sig(
    const float* __restrict__ xc, const float* __restrict__ w,
    const float* __restrict__ b, float* __restrict__ out)
{
    const int l = threadIdx.x, c = blockIdx.x, n = blockIdx.y;
    const float* ib = xc + (size_t)n * 128 * 128;
    float acc = b[c];
    for (int ic = 0; ic < 128; ++ic) acc = fmaf(w[c * 128 + ic], ib[ic * 128 + l], acc);
    out[(size_t)n * 65920 + 65536 + c * 128 + l] = 1.f / (1.f + expf(-0.01f * acc));
}

// ---------------------------------------------------------------------------
// KNN over the deterministic (l,a,b) grid, stable tie-break (== jnp.argsort)
// ---------------------------------------------------------------------------
__global__ void knn_kernel(int* __restrict__ knn)
{
    const int m = blockIdx.x * 256 + threadIdx.x;
    if (m >= 2048) return;
    const int l = m >> 4, a = (m >> 2) & 3, bb = m & 3;
    int bd[4] = {0x7fffffff, 0x7fffffff, 0x7fffffff, 0x7fffffff};
    int bi[4] = {0, 0, 0, 0};
    for (int j = 0; j < 2048; ++j) {
        const int dl = l - (j >> 4), da = a - ((j >> 2) & 3), db = bb - (j & 3);
        const int d = dl * dl + da * da + db * db;
        if (d < bd[3]) {
            bd[3] = d; bi[3] = j;
            #pragma unroll
            for (int p = 3; p > 0; --p)
                if (bd[p] < bd[p - 1]) {
                    int td = bd[p]; bd[p] = bd[p - 1]; bd[p - 1] = td;
                    int ti = bi[p]; bi[p] = bi[p - 1]; bi[p - 1] = ti;
                }
        }
    }
    #pragma unroll
    for (int k = 0; k < 4; ++k) knn[m * 4 + k] = bi[k];
}

// ---------------------------------------------------------------------------
// pos MLP (batch-independent): rel(3) -> 128 relu -> 128.  grid=8192 (m,k), block=128
// ---------------------------------------------------------------------------
__global__ __launch_bounds__(128) void pos_kernel(
    const int* __restrict__ knn, const float* __restrict__ d1w,
    const float* __restrict__ d1b, const float* __restrict__ d2w,
    const float* __restrict__ d2b, float* __restrict__ pos)
{
    __shared__ float hid[128];
    const int mk = blockIdx.x;
    const int m = mk >> 2;
    const int j = knn[mk];
    const float r0 = (float)((m >> 4) - (j >> 4));
    const float r1 = (float)(((m >> 2) & 3) - ((j >> 2) & 3));
    const float r2 = (float)((m & 3) - (j & 3));
    const int t = threadIdx.x;
    float h = d1b[t] + d1w[t * 3] * r0 + d1w[t * 3 + 1] * r1 + d1w[t * 3 + 2] * r2;
    hid[t] = fmaxf(h, 0.f);
    __syncthreads();
    float acc = d2b[t];
    for (int c = 0; c < 128; ++c) acc = fmaf(d2w[t * 128 + c], hid[c], acc);
    pos[(size_t)mk * 128 + t] = acc;
}

// ---------------------------------------------------------------------------
// x1 = feats@fc1.T+b ; q/k/v = x1@w*.T   grid=(2048,N), block=128
// ---------------------------------------------------------------------------
__global__ __launch_bounds__(128) void qkv_kernel(
    const float* __restrict__ xc, const float* __restrict__ fc1w,
    const float* __restrict__ fc1b, const float* __restrict__ wq,
    const float* __restrict__ wk, const float* __restrict__ wv,
    float* __restrict__ qb, float* __restrict__ kb, float* __restrict__ vb)
{
    __shared__ float feat[128], x1s[128];
    const int m = blockIdx.x, n = blockIdx.y, t = threadIdx.x;
    const int l = m >> 4;
    feat[t] = xc[((size_t)n * 128 + t) * 128 + l];
    __syncthreads();
    float acc = fc1b[t];
    for (int c = 0; c < 128; ++c) acc = fmaf(fc1w[t * 128 + c], feat[c], acc);
    x1s[t] = acc;
    __syncthreads();
    float q = 0.f, kk = 0.f, v = 0.f;
    for (int c = 0; c < 128; ++c) {
        const float xv = x1s[c];
        q  = fmaf(wq[t * 128 + c], xv, q);
        kk = fmaf(wk[t * 128 + c], xv, kk);
        v  = fmaf(wv[t * 128 + c], xv, v);
    }
    const size_t o = ((size_t)n * 2048 + m) * 128 + t;
    qb[o] = q; kb[o] = kk; vb[o] = v;
}

// ---------------------------------------------------------------------------
// attention block.  grid=(2048,N), block=128
// ---------------------------------------------------------------------------
__global__ __launch_bounds__(128) void attn_kernel(
    const float* __restrict__ xc, const float* __restrict__ qb,
    const float* __restrict__ kb, const float* __restrict__ vb,
    const int* __restrict__ knn, const float* __restrict__ pos,
    const float* __restrict__ g1w, const float* __restrict__ g1b,
    const float* __restrict__ g2w, const float* __restrict__ g2b,
    const float* __restrict__ fc2w, const float* __restrict__ fc2b,
    float* __restrict__ res2)
{
    __shared__ float As[4][128], Vp[4][128], Hs[4][128], Rs[128];
    const int m = blockIdx.x, n = blockIdx.y, t = threadIdx.x;
    const size_t base = ((size_t)n * 2048 + m) * 128;
    const float qv = qb[base + t];
    #pragma unroll
    for (int k = 0; k < 4; ++k) {
        const int j = knn[m * 4 + k];
        const float pv = pos[((size_t)(m * 4 + k)) * 128 + t];
        const size_t jo = ((size_t)n * 2048 + j) * 128 + t;
        As[k][t] = qv - kb[jo] + pv;
        Vp[k][t] = vb[jo] + pv;
    }
    __syncthreads();
    #pragma unroll
    for (int k = 0; k < 4; ++k) {
        float h = g1b[t];
        for (int c = 0; c < 128; ++c) h = fmaf(g1w[t * 128 + c], As[k][c], h);
        Hs[k][t] = fmaxf(h, 0.f);
    }
    __syncthreads();
    float a2[4];
    #pragma unroll
    for (int k = 0; k < 4; ++k) {
        float h = g2b[t];
        for (int c = 0; c < 128; ++c) h = fmaf(g2w[t * 128 + c], Hs[k][c], h);
        a2[k] = h * 0.08838834764831845f; // 1/sqrt(128)
    }
    const float mx = fmaxf(fmaxf(a2[0], a2[1]), fmaxf(a2[2], a2[3]));
    float e[4], s = 0.f;
    #pragma unroll
    for (int k = 0; k < 4; ++k) { e[k] = expf(a2[k] - mx); s += e[k]; }
    const float inv = 1.f / s;
    float r = 0.f;
    #pragma unroll
    for (int k = 0; k < 4; ++k) r = fmaf(e[k] * inv, Vp[k][t], r);
    Rs[t] = r;
    __syncthreads();
    float o = fc2b[t];
    for (int c = 0; c < 128; ++c) o = fmaf(fc2w[t * 128 + c], Rs[c], o);
    o += xc[((size_t)n * 128 + t) * 128 + (m >> 4)];
    res2[base + t] = o;
}

// ---------------------------------------------------------------------------
// xm = xc + mean over 16 expand positions.  32768 threads
// ---------------------------------------------------------------------------
__global__ void xm_kernel(const float* __restrict__ xc,
                          const float* __restrict__ res2, float* __restrict__ xm)
{
    const int idx = blockIdx.x * 256 + threadIdx.x;  // n*16384 + c*128 + l
    const int n = idx >> 14;
    const int r = idx & 16383;
    const int c = r >> 7, l = r & 127;
    float s = 0.f;
    const size_t base = ((size_t)n * 2048 + l * 16) * 128 + c;
    #pragma unroll
    for (int e = 0; e < 16; ++e) s += res2[base + (size_t)e * 128];
    xm[idx] = xc[idx] + s * 0.0625f;
}

// ---------------------------------------------------------------------------
// Stage A: T[(n*512+o), t*32+s] = sum_c xm[n,c,t]*c3d_w[o,c,s]  -> bf16
// ---------------------------------------------------------------------------
__global__ void stageA_kernel(const float* __restrict__ xm,
                              const float* __restrict__ c3dw, bf16_t* __restrict__ T)
{
    const int idx = blockIdx.x * 256 + threadIdx.x;  // 0 .. 1024*4096-1
    const int row = idx >> 12;
    const int kk = idx & 4095;
    const int tt = kk >> 5, s = kk & 31;
    const int n = row >> 9, o = row & 511;
    float acc = 0.f;
    const float* xb = xm + (size_t)n * 16384 + tt;    // stride 128 over c
    const float* wb = c3dw + (size_t)o * 128 * 32 + s; // stride 32 over c
    for (int c = 0; c < 128; ++c) acc = fmaf(xb[c * 128], wb[c * 32], acc);
    T[idx] = f2bf(acc);
}

// ---------------------------------------------------------------------------
// f32 -> bf16, 4 elements per thread
// ---------------------------------------------------------------------------
__global__ void f32_to_bf16_x4(const float* __restrict__ in,
                               bf16_t* __restrict__ out, int n4)
{
    const int i = blockIdx.x * 256 + threadIdx.x;
    if (i >= n4) return;
    const float4 v = ((const float4*)in)[i];
    v4bf o;
    o[0] = f2bf(v.x); o[1] = f2bf(v.y); o[2] = f2bf(v.z); o[3] = f2bf(v.w);
    ((v4bf*)out)[i] = o;
}

// ---------------------------------------------------------------------------
// bf16 WMMA GEMM: C = relu(A(MxK) * B(KxN) + bias[row]), bf16 out, fp32 acc.
// block = 256 (8 waves). Block tile 64(M) x 256(N); wave tile 32x64 (8 WMMA/kstep).
// M,N multiples of 64/256, K multiple of 32. grid=(N/256, M/64, batch)
// ---------------------------------------------------------------------------
__global__ __launch_bounds__(256) void gemm_bf16_wmma(
    const bf16_t* __restrict__ A, const bf16_t* __restrict__ B,
    const float* __restrict__ bias, bf16_t* __restrict__ C,
    int K, int lda, int ldb, int ldc, int biasMask,
    long long sA, long long sB, long long sC)
{
    const int tid  = threadIdx.x;
    const int wave = tid >> 5;
    const int lane = tid & 31;
    const int half = lane >> 4;
    const int lid  = lane & 15;
    const int waveM = wave >> 2;     // 0..1
    const int waveN = wave & 3;      // 0..3
    const int tileN = blockIdx.x * 256 + waveN * 64;
    const int tileM = blockIdx.y * 64 + waveM * 32;
    const long long bz = blockIdx.z;
    A += bz * sA; B += bz * sB; C += bz * sC;

    v8f acc[2][4];
    #pragma unroll
    for (int i = 0; i < 2; ++i)
        #pragma unroll
        for (int j = 0; j < 4; ++j)
            #pragma unroll
            for (int r = 0; r < 8; ++r) acc[i][j][r] = 0.f;

    for (int k0 = 0; k0 < K; k0 += 32) {
        // A fragments: lane<16 -> K {0..7,16..23}; lane>=16 -> K {8..15,24..31}
        v16bf af[2];
        #pragma unroll
        for (int i = 0; i < 2; ++i) {
            const bf16_t* ap = A + (size_t)(tileM + i * 16 + lid) * (size_t)lda + k0 + half * 8;
            v8bf lo = *(const v8bf*)ap;
            v8bf hi = *(const v8bf*)(ap + 16);
            #pragma unroll
            for (int e = 0; e < 8; ++e) { af[i][e] = lo[e]; af[i][e + 8] = hi[e]; }
        }
        // B fragments: lane -> column; lane<16 -> K {0..7,16..23}; lane>=16 -> K {8..15,24..31}
        v16bf bfr[4];
        #pragma unroll
        for (int j = 0; j < 4; ++j) {
            const bf16_t* bp = B + (size_t)(k0 + half * 8) * (size_t)ldb + tileN + j * 16 + lid;
            #pragma unroll
            for (int e = 0; e < 16; ++e) bfr[j][e] = bp[(size_t)(e + (e >> 3) * 8) * (size_t)ldb];
        }
        #pragma unroll
        for (int i = 0; i < 2; ++i)
            #pragma unroll
            for (int j = 0; j < 4; ++j)
                acc[i][j] = __builtin_amdgcn_wmma_f32_16x16x32_bf16(
                    false, af[i], false, bfr[j], (short)0, acc[i][j], false, false);
    }

    // C layout: VGPR r -> row half*8+r; lane -> col
    #pragma unroll
    for (int i = 0; i < 2; ++i)
        #pragma unroll
        for (int j = 0; j < 4; ++j)
            #pragma unroll
            for (int r = 0; r < 8; ++r) {
                const int row = tileM + i * 16 + half * 8 + r;
                const int col = tileN + j * 16 + lid;
                const float v = acc[i][j][r] + bias[row & biasMask];
                C[(size_t)row * (size_t)ldc + col] = f2bf(fmaxf(v, 0.f));
            }
}

// ---------------------------------------------------------------------------
// 3x3 conv, pad 1, bf16 in/out, fp32 weights, ReLU. grid=(64, Cout, N), block=256
// ---------------------------------------------------------------------------
__global__ __launch_bounds__(256) void conv3x3_bf16_relu(
    const bf16_t* __restrict__ in, const float* __restrict__ w,
    const float* __restrict__ bias, bf16_t* __restrict__ out, int Cin)
{
    __shared__ float ws[1152];
    const int oc = blockIdx.y;
    const int n  = blockIdx.z;
    const int nw = Cin * 9;
    for (int i = threadIdx.x; i < nw; i += 256) ws[i] = w[(size_t)oc * nw + i];
    __syncthreads();
    const int xy = blockIdx.x * 256 + threadIdx.x;
    const int x = xy >> 7, y = xy & 127;
    const bf16_t* inb = in + (size_t)n * Cin * 16384;
    float acc = bias[oc];
    for (int ic = 0; ic < Cin; ++ic) {
        const bf16_t* chan = inb + (size_t)ic * 16384;
        const float* wk = ws + ic * 9;
        #pragma unroll
        for (int dx = -1; dx <= 1; ++dx) {
            const int xx = x + dx;
            if (xx < 0 || xx > 127) continue;
            const bf16_t* row = chan + xx * 128;
            #pragma unroll
            for (int dy = -1; dy <= 1; ++dy) {
                const int yy = y + dy;
                if (yy < 0 || yy > 127) continue;
                acc = fmaf(wk[(dx + 1) * 3 + (dy + 1)], bf2f(row[yy]), acc);
            }
        }
    }
    out[(size_t)(n * gridDim.y + oc) * 16384 + xy] = f2bf(fmaxf(acc, 0.f));
}

// ---------------------------------------------------------------------------
// p4: 1x1 conv 64->4 + sigmoid -> d_out.  131072 threads
// ---------------------------------------------------------------------------
__global__ void p4_sigmoid(const bf16_t* __restrict__ in, const float* __restrict__ w,
                           const float* __restrict__ b, float* __restrict__ out)
{
    const int idx = blockIdx.x * 256 + threadIdx.x;
    const int n = idx >> 16;
    const int r = idx & 65535;
    const int o = r >> 14;
    const int xy = r & 16383;
    const bf16_t* inb = in + (size_t)n * 64 * 16384 + xy;
    float acc = b[o];
    for (int ic = 0; ic < 64; ++ic) acc = fmaf(w[o * 64 + ic], bf2f(inb[(size_t)ic * 16384]), acc);
    out[(size_t)n * 65920 + o * 16384 + xy] = 1.f / (1.f + expf(-acc));
}

// ---------------------------------------------------------------------------
extern "C" void kernel_launch(void* const* d_in, const int* in_sizes, int n_in,
                              void* d_out, int out_size, void* d_ws, size_t ws_size,
                              hipStream_t stream) {
    (void)in_sizes; (void)n_in; (void)out_size; (void)ws_size;
    const float* x      = (const float*)d_in[0];
    const float* conv1w = (const float*)d_in[1];
    const float* conv1b = (const float*)d_in[2];
    const float* conv2w = (const float*)d_in[3];
    const float* conv2b = (const float*)d_in[4];
    const float* conv3w = (const float*)d_in[5];
    const float* conv3b = (const float*)d_in[6];
    const float* fc1w   = (const float*)d_in[7];
    const float* fc1b   = (const float*)d_in[8];
    const float* fc2w   = (const float*)d_in[9];
    const float* fc2b   = (const float*)d_in[10];
    const float* d1w    = (const float*)d_in[11];
    const float* d1b    = (const float*)d_in[12];
    const float* d2w    = (const float*)d_in[13];
    const float* d2b    = (const float*)d_in[14];
    const float* g1w    = (const float*)d_in[15];
    const float* g1b    = (const float*)d_in[16];
    const float* g2w    = (const float*)d_in[17];
    const float* g2b    = (const float*)d_in[18];
    const float* wq     = (const float*)d_in[19];
    const float* wk     = (const float*)d_in[20];
    const float* wv     = (const float*)d_in[21];
    const float* c3dw   = (const float*)d_in[22];
    const float* c3db   = (const float*)d_in[23];
    const float* p1w    = (const float*)d_in[24];
    const float* p1b    = (const float*)d_in[25];
    const float* p2w    = (const float*)d_in[26];
    const float* p2b    = (const float*)d_in[27];
    const float* p3w    = (const float*)d_in[28];
    const float* p3b    = (const float*)d_in[29];
    const float* p4w    = (const float*)d_in[30];
    const float* p4b    = (const float*)d_in[31];
    const float* mask   = (const float*)d_in[32];
    float* out = (float*)d_out;

    char* base = (char*)d_ws;
    size_t off = 0;
    auto alloc = [&](size_t bytes) -> void* {
        void* p = base + off;
        off += (bytes + 255) & ~(size_t)255;
        return p;
    };
    float*  h1    = (float*) alloc((size_t)2 * 512 * 128 * 4);
    float*  xc    = (float*) alloc((size_t)2 * 128 * 128 * 4);
    int*    knn   = (int*)   alloc((size_t)2048 * 4 * 4);
    float*  pos   = (float*) alloc((size_t)2048 * 4 * 128 * 4);
    float*  qb    = (float*) alloc((size_t)2 * 2048 * 128 * 4);
    float*  kb    = (float*) alloc((size_t)2 * 2048 * 128 * 4);
    float*  vb    = (float*) alloc((size_t)2 * 2048 * 128 * 4);
    float*  res2  = (float*) alloc((size_t)2 * 2048 * 128 * 4);
    float*  xm    = (float*) alloc((size_t)2 * 128 * 128 * 4);
    bf16_t* T     = (bf16_t*)alloc((size_t)1024 * 4096 * 2);
    bf16_t* maskb = (bf16_t*)alloc((size_t)4096 * 16384 * 2);
    bf16_t* h3    = (bf16_t*)alloc((size_t)2 * 512 * 16384 * 2);
    bf16_t* p1wb  = (bf16_t*)alloc((size_t)128 * 512 * 2);
    bf16_t* p1o   = (bf16_t*)alloc((size_t)2 * 128 * 16384 * 2);
    bf16_t* p2o   = (bf16_t*)alloc((size_t)2 * 64 * 16384 * 2);
    bf16_t* p3o   = (bf16_t*)alloc((size_t)2 * 64 * 16384 * 2);

    // head
    conv1d_k3_relu<<<dim3(512, 2), 128, 0, stream>>>(x,  conv1w, conv1b, h1, 200, 400 * 128, 512);
    conv1d_k3_relu<<<dim3(128, 2), 128, 0, stream>>>(h1, conv2w, conv2b, xc, 512, 512 * 128, 128);
    conv3_sig<<<dim3(3, 2), 128, 0, stream>>>(xc, conv3w, conv3b, out);

    // point transformer block
    knn_kernel<<<8, 256, 0, stream>>>(knn);
    pos_kernel<<<8192, 128, 0, stream>>>(knn, d1w, d1b, d2w, d2b, pos);
    qkv_kernel<<<dim3(2048, 2), 128, 0, stream>>>(xc, fc1w, fc1b, wq, wk, wv, qb, kb, vb);
    attn_kernel<<<dim3(2048, 2), 128, 0, stream>>>(xc, qb, kb, vb, knn, pos,
                                                   g1w, g1b, g2w, g2b, fc2w, fc2b, res2);
    xm_kernel<<<128, 256, 0, stream>>>(xc, res2, xm);

    // reordered mask contraction:  T (c-contraction) then big WMMA GEMM over (t,s)
    stageA_kernel<<<16384, 256, 0, stream>>>(xm, c3dw, T);
    f32_to_bf16_x4<<<65536, 256, 0, stream>>>(mask, maskb, 16777216);
    f32_to_bf16_x4<<<64, 256, 0, stream>>>(p1w, p1wb, 16384);

    // h3[n,o,xy] = relu(T(1024x4096) @ maskb(4096x16384) + c3d_b)   (137 GFLOP, L2-resident B)
    gemm_bf16_wmma<<<dim3(64, 16, 1), 256, 0, stream>>>(T, maskb, c3db, h3,
                                                        4096, 4096, 16384, 16384, 511, 0, 0, 0);
    // p1: 1x1 conv as GEMM per batch
    gemm_bf16_wmma<<<dim3(64, 2, 2), 256, 0, stream>>>(p1wb, h3, p1b, p1o,
                                                       512, 512, 16384, 16384, 127,
                                                       0, (long long)512 * 16384, (long long)128 * 16384);
    // p2 / p3 direct convs
    conv3x3_bf16_relu<<<dim3(64, 64, 2), 256, 0, stream>>>(p1o, p2w, p2b, p2o, 128);
    conv3x3_bf16_relu<<<dim3(64, 64, 2), 256, 0, stream>>>(p2o, p3w, p3b, p3o, 64);
    // p4 + sigmoid -> d_out
    p4_sigmoid<<<512, 256, 0, stream>>>(p3o, p4w, p4b, out);
}
